// DGM_d_5248450035895
// MI455X (gfx1250) — matrix operand order repaired and checked by reading, not compiled, not run on hardware
//
#include <hip/hip_runtime.h>
#include <hip/hip_bf16.h>

typedef __attribute__((ext_vector_type(16))) __bf16 v16bf;
typedef __attribute__((ext_vector_type(8)))  __bf16 v8bf;
typedef __attribute__((ext_vector_type(8)))  float  v8f;

#define BATCH 4
#define NPTS  4096
#define DIM   64
#define KNN   8
#define JTILES (NPTS / 16)
#define WPB   4   // waves per block in knn kernel

// ---------------------------------------------------------------------------
// Kernel 1: 3-layer MLP (torch Linear: y = x @ W^T + b), plus per-row ||h||^2
// and bf16 hi/lo split of h for split-precision WMMA Gram.
// One block (64 threads) per row; thread t computes output feature t.
// ---------------------------------------------------------------------------
__global__ __launch_bounds__(64) void mlp_kernel(
    const float* __restrict__ x,
    const float* __restrict__ W1, const float* __restrict__ b1,
    const float* __restrict__ W2, const float* __restrict__ b2,
    const float* __restrict__ W3, const float* __restrict__ b3,
    float* __restrict__ h, __bf16* __restrict__ hhi, __bf16* __restrict__ hlo,
    float* __restrict__ sq)
{
    __shared__ float buf0[DIM];
    __shared__ float buf1[DIM];
    const int row = blockIdx.x;          // 0 .. BATCH*NPTS-1
    const int t   = threadIdx.x;         // 0 .. 63

    buf0[t] = x[(size_t)row * DIM + t];
    __syncthreads();

    float acc = b1[t];
    #pragma unroll 8
    for (int k = 0; k < DIM; ++k) acc += W1[t * DIM + k] * buf0[k];
    buf1[t] = fmaxf(acc, 0.0f);
    __syncthreads();

    acc = b2[t];
    #pragma unroll 8
    for (int k = 0; k < DIM; ++k) acc += W2[t * DIM + k] * buf1[k];
    buf0[t] = fmaxf(acc, 0.0f);
    __syncthreads();

    acc = b3[t];
    #pragma unroll 8
    for (int k = 0; k < DIM; ++k) acc += W3[t * DIM + k] * buf0[k];

    const float hv = acc;
    h[(size_t)row * DIM + t] = hv;
    __bf16 hi = (__bf16)hv;                 // RNE
    __bf16 lo = (__bf16)(hv - (float)hi);   // residual
    hhi[(size_t)row * DIM + t] = hi;
    hlo[(size_t)row * DIM + t] = lo;

    buf1[t] = hv * hv;
    __syncthreads();
    if (t == 0) {
        float s = 0.0f;
        #pragma unroll 8
        for (int i = 0; i < DIM; ++i) s += buf1[i];
        sq[row] = s;
    }
}

// ---------------------------------------------------------------------------
// WMMA fragment loads (bf16 16x16x32, wave32; contiguous k order, same for A and B)
// A (16x32, MxK): lane L holds row base+(L&15); elems e = K kbase+16*(L>>4)+e.
// B (32x16, KxN): lane L holds col base+(L&15); elems e = K kbase+16*(L>>4)+e.
// ---------------------------------------------------------------------------
__device__ __forceinline__ v16bf load_a_frag(const __bf16* rowp, int kbase, int khalf) {
    return *reinterpret_cast<const v16bf*>(rowp + kbase + 16 * khalf);
}

__device__ __forceinline__ v16bf load_b_frag(const __bf16* rowp, int kbase, int khalf) {
    return *reinterpret_cast<const v16bf*>(rowp + kbase + 16 * khalf);
}

// ---------------------------------------------------------------------------
// Kernel 2: fused Gram (split-bf16 WMMA) + per-lane register top-8 + merge +
// exact f32 rescoring.
//
// Tile roles: D = H[j-tile](A, streamed) x H[i-tile]^T(B, fixed). With the C/D
// layout, lane L's 8 accumulator elements are 8 candidates j = j0+8*(L>>4)+v
// for the single row i = i0+(L&15). Hot loop: 8 key FMAs + 7 v_min + ONE
// cutoff test; the full insertion network lives in a rarely-taken cold block.
// ---------------------------------------------------------------------------
__global__ __launch_bounds__(WPB * 32) void knn_topk_kernel(
    const float* __restrict__ h,
    const __bf16* __restrict__ hhi, const __bf16* __restrict__ hlo,
    const float* __restrict__ sq,
    const float* __restrict__ temperature,
    float* __restrict__ edges,   // [BATCH][NPTS*KNN][2] as float
    float* __restrict__ logp)    // [BATCH][NPTS][KNN]
{
    __shared__ float skey[WPB][16 * 17];   // stride 17: conflict-free
    __shared__ int   sidx[WPB][16 * 17];

    const int wave  = threadIdx.x >> 5;
    const int lane  = threadIdx.x & 31;
    const int r     = lane & 15;           // row-in-i-tile this lane serves
    const int khalf = lane >> 4;           // which 8-candidate half per tile

    const int T  = blockIdx.x * WPB + wave;   // 0 .. BATCH*JTILES-1
    const int bi = T >> 8;                    // / (NPTS/16)
    const int i0 = (T & (JTILES - 1)) << 4;

    const __bf16* Hhi = hhi + (size_t)bi * NPTS * DIM;
    const __bf16* Hlo = hlo + (size_t)bi * NPTS * DIM;
    const float*  SQ  = sq  + (size_t)bi * NPTS;

    // Fixed B fragments: this wave's 16 i-rows as WMMA B (columns), hi and lo.
    const __bf16* brow_hi = Hhi + (size_t)(i0 + r) * DIM;
    const __bf16* brow_lo = Hlo + (size_t)(i0 + r) * DIM;
    const v16bf bihi0 = load_b_frag(brow_hi, 0, khalf);
    const v16bf bihi1 = load_b_frag(brow_hi, 32, khalf);
    const v16bf bilo0 = load_b_frag(brow_lo, 0, khalf);
    const v16bf bilo1 = load_b_frag(brow_lo, 32, khalf);

    // per-lane sorted top-8 (ascending key), candidates arrive j-ascending
    float bd[KNN];
    int   bj[KNN];
    #pragma unroll
    for (int p = 0; p < KNN; ++p) { bd[p] = 3.4e38f; bj[p] = 0x7fffffff; }

    for (int jt = 0; jt < JTILES; ++jt) {
        const int j0 = jt << 4;
        const __bf16* arow_hi = Hhi + (size_t)(j0 + r) * DIM;
        const __bf16* arow_lo = Hlo + (size_t)(j0 + r) * DIM;
        const v16bf ajhi0 = load_a_frag(arow_hi, 0, khalf);
        const v16bf ajhi1 = load_a_frag(arow_hi, 32, khalf);
        const v16bf ajlo0 = load_a_frag(arow_lo, 0, khalf);
        const v16bf ajlo1 = load_a_frag(arow_lo, 32, khalf);
        const v8f   sqj8  = *reinterpret_cast<const v8f*>(SQ + j0 + 8 * khalf);

        // Gram tile: hi*hi + hi*lo + lo*hi (f32 accumulate), K=64 -> 2 wmma/term
        v8f c = {};
        c = __builtin_amdgcn_wmma_f32_16x16x32_bf16(false, ajhi0, false, bihi0, (short)0, c, false, false);
        c = __builtin_amdgcn_wmma_f32_16x16x32_bf16(false, ajhi1, false, bihi1, (short)0, c, false, false);
        c = __builtin_amdgcn_wmma_f32_16x16x32_bf16(false, ajhi0, false, bilo0, (short)0, c, false, false);
        c = __builtin_amdgcn_wmma_f32_16x16x32_bf16(false, ajhi1, false, bilo1, (short)0, c, false, false);
        c = __builtin_amdgcn_wmma_f32_16x16x32_bf16(false, ajlo0, false, bihi0, (short)0, c, false, false);
        c = __builtin_amdgcn_wmma_f32_16x16x32_bf16(false, ajlo1, false, bihi1, (short)0, c, false, false);

        // keys: candidate j = j0 + 8*khalf + v for row i = i0 + r.
        // key = sq_j - 2*<h_i,h_j> (row-constant sq_i / monotone tau dropped).
        float dc[8];
        #pragma unroll
        for (int v = 0; v < 8; ++v) dc[v] = __builtin_fmaf(-2.0f, c[v], sqj8[v]);

        // hot filter: one cutoff test per 8 candidates
        const float m01 = fminf(dc[0], dc[1]);
        const float m23 = fminf(dc[2], dc[3]);
        const float m45 = fminf(dc[4], dc[5]);
        const float m67 = fminf(dc[6], dc[7]);
        const float kmin = fminf(fminf(m01, m23), fminf(m45, m67));

        if (kmin < bd[KNN - 1]) {            // cold: some candidate qualifies
            #pragma unroll
            for (int v = 0; v < 8; ++v) {
                if (dc[v] < bd[KNN - 1]) {   // strict: j ascending => stable
                    bd[KNN - 1] = dc[v];
                    bj[KNN - 1] = j0 + 8 * khalf + v;
                    #pragma unroll
                    for (int p = KNN - 1; p > 0; --p) {
                        const bool sw = bd[p] < bd[p - 1];
                        const float dlo = sw ? bd[p] : bd[p - 1];
                        const float dhi = sw ? bd[p - 1] : bd[p];
                        const int   jlo = sw ? bj[p] : bj[p - 1];
                        const int   jhi = sw ? bj[p - 1] : bj[p];
                        bd[p - 1] = dlo; bd[p] = dhi;
                        bj[p - 1] = jlo; bj[p] = jhi;
                    }
                }
            }
        }
    }

    // Publish both half-lists per row, then lanes 0..15 merge 16 -> 8 with the
    // full lexicographic (key, j) rule (== stable jax top_k ordering).
    #pragma unroll
    for (int p = 0; p < KNN; ++p) {
        const int s = khalf * KNN + p;
        skey[wave][r * 17 + s] = bd[p];
        sidx[wave][r * 17 + s] = bj[p];
    }
    asm volatile("s_wait_dscnt 0" ::: "memory");

    if (lane < 16) {
        float fd[KNN];
        int   fj[KNN];
        #pragma unroll
        for (int p = 0; p < KNN; ++p) { fd[p] = 3.4e38f; fj[p] = 0x7fffffff; }
        #pragma unroll 1
        for (int s = 0; s < 16; ++s) {
            const float dcand = skey[wave][lane * 17 + s];
            const int   jcand = sidx[wave][lane * 17 + s];
            if (dcand < fd[KNN - 1] ||
                (dcand == fd[KNN - 1] && jcand < fj[KNN - 1])) {
                fd[KNN - 1] = dcand; fj[KNN - 1] = jcand;
                #pragma unroll
                for (int p = KNN - 1; p > 0; --p) {
                    const bool sw = (fd[p] < fd[p - 1]) ||
                                    (fd[p] == fd[p - 1] && fj[p] < fj[p - 1]);
                    const float dlo = sw ? fd[p] : fd[p - 1];
                    const float dhi = sw ? fd[p - 1] : fd[p];
                    const int   jlo = sw ? fj[p] : fj[p - 1];
                    const int   jhi = sw ? fj[p - 1] : fj[p];
                    fd[p - 1] = dlo; fd[p] = dhi;
                    fj[p - 1] = jlo; fj[p] = jhi;
                }
            }
        }
        // Republish final indices so the rescoring loop can index dynamically.
        #pragma unroll
        for (int p = 0; p < KNN; ++p) sidx[wave][lane * 17 + p] = fj[p];
    }
    asm volatile("s_wait_dscnt 0" ::: "memory");

    // Exact f32 re-scoring (matches reference's dng recompute) + edge list.
    if (lane < 16) {
        const int i = i0 + lane;
        const float tc  = fminf(fmaxf(*temperature, -5.0f), 5.0f);
        const float tau = expf(tc);
        const float4* Hi4 =
            reinterpret_cast<const float4*>(h + ((size_t)bi * NPTS + i) * DIM);
        #pragma unroll 1
        for (int kk = 0; kk < KNN; ++kk) {
            const int j = sidx[wave][lane * 17 + kk];
            const float4* Hj4 =
                reinterpret_cast<const float4*>(h + ((size_t)bi * NPTS + j) * DIM);
            float s = 0.0f;
            #pragma unroll 4
            for (int t = 0; t < DIM / 4; ++t) {
                const float4 a = Hi4[t];
                const float4 bv = Hj4[t];
                const float d0 = bv.x - a.x, d1 = bv.y - a.y;
                const float d2 = bv.z - a.z, d3 = bv.w - a.w;
                s += d0 * d0 + d1 * d1 + d2 * d2 + d3 * d3;
            }
            const size_t p = ((size_t)bi * NPTS + i) * KNN + kk;
            logp[p]          = -s * tau;
            edges[2 * p]     = (float)j;   // idx_flat
            edges[2 * p + 1] = (float)i;   // rows
        }
    }
}

// ---------------------------------------------------------------------------
// Launch: inputs in setup_inputs order:
// 0:x 1:W1 2:b1 3:W2 4:b2 5:W3 6:b3 7:temperature 8:k
// Output: edges (BATCH*NPTS*KNN*2) then logprobs (BATCH*NPTS*KNN), as float.
// ---------------------------------------------------------------------------
extern "C" void kernel_launch(void* const* d_in, const int* in_sizes, int n_in,
                              void* d_out, int out_size, void* d_ws, size_t ws_size,
                              hipStream_t stream) {
    const float* x  = (const float*)d_in[0];
    const float* W1 = (const float*)d_in[1];
    const float* b1 = (const float*)d_in[2];
    const float* W2 = (const float*)d_in[3];
    const float* b2 = (const float*)d_in[4];
    const float* W3 = (const float*)d_in[5];
    const float* b3 = (const float*)d_in[6];
    const float* temperature = (const float*)d_in[7];
    (void)in_sizes; (void)n_in; (void)ws_size;

    // workspace layout
    char* w = (char*)d_ws;
    float*  h   = (float*)w;                                          // 4 MB
    __bf16* hhi = (__bf16*)(w + (size_t)BATCH * NPTS * DIM * 4);      // 2 MB
    __bf16* hlo = (__bf16*)(w + (size_t)BATCH * NPTS * DIM * 6);      // 2 MB
    float*  sqv = (float*)(w + (size_t)BATCH * NPTS * DIM * 8);       // 64 KB

    float* edges = (float*)d_out;
    float* logp  = (float*)d_out + (size_t)BATCH * NPTS * KNN * 2;
    (void)out_size;

    mlp_kernel<<<BATCH * NPTS, DIM, 0, stream>>>(x, W1, b1, W2, b2, W3, b3,
                                                 h, hhi, hlo, sqv);

    const int tiles  = BATCH * JTILES;     // 1024 wave-tiles
    const int blocks = tiles / WPB;        // 256 blocks of 4 waves
    knn_topk_kernel<<<blocks, WPB * 32, 0, stream>>>(h, hhi, hlo, sqv,
                                                     temperature, edges, logp);
}
